// JointEntityRelation_11141145166426
// MI455X (gfx1250) — matrix operand order repaired and checked by reading, not compiled, not run on hardware
//
#include <hip/hip_runtime.h>
#include <hip/hip_bf16.h>
#include <cstdint>
#include <cstddef>

// Problem constants
#define BB 8
#define SS 128
#define HH 768
#define EE 4
#define RR 13
#define DD (HH + EE)   // 772

// Padded LDS row strides (elements) — chosen so row stride in dwords is
// != 0 mod 64 banks, keeping ds_load_b128 fragment reads conflict-free,
// while staying 16B-aligned.
#define ASTR  776      // As rows:  776*2B = 1552B = 388 dwords (≡4 mod 64)
#define A2STR 136      // A2 rows:  136*2B =  272B =  68 dwords (≡4 mod 64)
#define BROW  72       // B tile rows: 72*2B = 144B = 36 dwords (≡36 mod 64)
#define BTILE_ELEMS (128 * BROW)          // one K=64 tile: 128 n-rows
#define BTILE_BYTES (BTILE_ELEMS * 2)     // 18432 B

#define AS_BYTES   (SS * ASTR * 2)                    // 198656
#define A2_OFF     AS_BYTES
#define A2_BYTES   (SS * A2STR * 2)                   //  34816
#define BS_OFF     (A2_OFF + A2_BYTES)                // 233472
#define BS_BYTES   (2 * BTILE_BYTES)                  //  36864
#define REL_OFF    (BS_OFF + BS_BYTES)                // 270336
#define SMEM_TOTAL (REL_OFF + SS * 16 * 4)            // 278528

typedef __bf16 bf16;
typedef __attribute__((ext_vector_type(16))) __bf16 v16bf;
typedef __attribute__((ext_vector_type(8)))  __bf16 v8bf;
typedef __attribute__((ext_vector_type(4)))  __bf16 v4bf;
typedef __attribute__((ext_vector_type(8)))  float  v8f;

union Frag16 { v16bf v; v8bf h[2]; };

__device__ __forceinline__ float mishf(float x) {
    float sp = (x > 15.0f) ? x : __logf(1.0f + __expf(x));
    return x * tanhf(sp);
}

// ---------------------------------------------------------------------------
// Stage 1a: entity head.  One block per (b,s) row.
// ---------------------------------------------------------------------------
__global__ void __launch_bounds__(256)
entity_kernel(const float* __restrict__ emb, const float* __restrict__ W1e,
              const float* __restrict__ b1e, const float* __restrict__ W2e,
              const float* __restrict__ b2e,
              float* __restrict__ entity_out, float* __restrict__ entlog_ws) {
    __shared__ float row[HH];
    __shared__ float logits[EE];
    const int tid = threadIdx.x;
    const int r = blockIdx.x;                 // b*S + s
    for (int k = tid; k < HH; k += 256) row[k] = emb[(size_t)r * HH + k];
    if (tid < EE) logits[tid] = 0.0f;
    __syncthreads();

    float part[EE] = {0.f, 0.f, 0.f, 0.f};
    for (int c = tid; c < HH; c += 256) {
        float acc = b1e[c];
        for (int k = 0; k < HH; ++k) acc += row[k] * W1e[(size_t)k * HH + c];
        float hv = mishf(acc);
        #pragma unroll
        for (int e = 0; e < EE; ++e) part[e] += hv * W2e[(size_t)c * EE + e];
    }
    #pragma unroll
    for (int e = 0; e < EE; ++e) atomicAdd(&logits[e], part[e]);
    __syncthreads();

    if (tid == 0) {
        float l[EE], mx = -1e30f;
        #pragma unroll
        for (int e = 0; e < EE; ++e) { l[e] = logits[e] + b2e[e]; mx = fmaxf(mx, l[e]); }
        float s = 0.f;
        #pragma unroll
        for (int e = 0; e < EE; ++e) s += __expf(l[e] - mx);
        float lse = mx + __logf(s);
        #pragma unroll
        for (int e = 0; e < EE; ++e) {
            entity_out[(size_t)r * EE + e] = l[e] - lse;
            entlog_ws[(size_t)r * EE + e]  = l[e];
        }
    }
}

// ---------------------------------------------------------------------------
// Stage 1b: pi = ee @ Wl[:D], pj = ee @ Wl[D:].
// ---------------------------------------------------------------------------
__global__ void __launch_bounds__(256)
pij_kernel(const float* __restrict__ emb, const float* __restrict__ entlog_ws,
           const float* __restrict__ Wl, float* __restrict__ pi, float* __restrict__ pj) {
    __shared__ float ee[DD];
    const int tid = threadIdx.x;
    const int r = blockIdx.x;
    const int which = blockIdx.y;             // 0 -> pi, 1 -> pj
    for (int k = tid; k < DD; k += 256)
        ee[k] = (k < HH) ? emb[(size_t)r * HH + k] : entlog_ws[(size_t)r * EE + (k - HH)];
    __syncthreads();

    const float* Wb = Wl + (which ? (size_t)DD * HH : (size_t)0);
    float* out = which ? pj : pi;
    for (int c = tid; c < HH; c += 256) {
        float acc = 0.f;
        for (int k = 0; k < DD; ++k) acc += ee[k] * Wb[(size_t)k * HH + c];
        out[(size_t)r * HH + c] = acc;
    }
}

// ---------------------------------------------------------------------------
// Stage 1c: weight conversion (transpose + bf16).
// ---------------------------------------------------------------------------
__global__ void conv_kernel(const float* __restrict__ W1r, const float* __restrict__ W2r,
                            bf16* __restrict__ W1rT, bf16* __restrict__ W2rT) {
    const int idx = blockIdx.x * 256 + threadIdx.x;
    const int t1 = HH * HH;
    if (idx < t1) {
        int n = idx / HH, k = idx - n * HH;
        W1rT[idx] = (bf16)W1r[(size_t)k * HH + n];
    } else if (idx < t1 + 16 * HH) {
        int i2 = idx - t1;
        int r = i2 / HH, n = i2 - r * HH;
        W2rT[i2] = (r < RR) ? (bf16)W2r[(size_t)n * RR + r] : (bf16)0.0f;
    }
}

// ---------------------------------------------------------------------------
// Async DMA: stage one 128(n) x 64(k) bf16 tile of W1rT into LDS.
// 16 KB of payload = 1024 x 16B chunks, 4 per thread, via
// global_load_async_to_lds_b128 (tracked by ASYNCcnt, no VGPR data).
// ---------------------------------------------------------------------------
__device__ __forceinline__ void
async_copy_b_tile(const bf16* __restrict__ W1rT, unsigned ldsTileAddr,
                  int nbase, int kb2, int tid) {
    unsigned long long sbase = (unsigned long long)(uintptr_t)W1rT;
    #pragma unroll
    for (int c = 0; c < 4; ++c) {
        int chunk = tid + c * 256;            // 0..1023
        int n = chunk >> 3;                   // 0..127  (row within tile)
        int q = chunk & 7;                    // 16B chunk within 128B row
        unsigned voff = (unsigned)((nbase + n) * (HH * 2) + kb2 * 2 + q * 16);
        unsigned ldst = ldsTileAddr + (unsigned)(n * (BROW * 2) + q * 16);
        asm volatile("global_load_async_to_lds_b128 %0, %1, %2"
                     :: "v"(ldst), "v"(voff), "s"(sbase)
                     : "memory");
    }
}

// ---------------------------------------------------------------------------
// Stage 2: fully fused relation head.  One WG per (b,i).
// ---------------------------------------------------------------------------
__global__ void __launch_bounds__(256)
rel_kernel(const float* __restrict__ pi, const float* __restrict__ pj,
           const float* __restrict__ bl, const bf16* __restrict__ W1rT,
           const float* __restrict__ b1r, const bf16* __restrict__ W2rT,
           const float* __restrict__ b2r, float* __restrict__ rel_out) {
    extern __shared__ char smem[];
    bf16*  As     = (bf16*)smem;                       // 128 x ASTR bf16
    bf16*  A2     = (bf16*)(smem + A2_OFF);            // 128 x A2STR bf16
    bf16*  Bs     = (bf16*)(smem + BS_OFF);            // 2 x (128 x BROW) bf16
    float* relbuf = (float*)(smem + REL_OFF);          // 128 x 16 f32

    const unsigned BsAddr = (unsigned)(uintptr_t)Bs;   // LDS byte address (addr[31:0])

    const int tid  = threadIdx.x;
    const int blk  = blockIdx.x;
    const int b    = blk >> 7;
    const int i    = blk & 127;
    const int w    = tid >> 5;        // wave id: row strip [16w, 16w+16)
    const int lane = tid & 31;
    const int hs   = (lane < 16) ? 0 : 1;
    const int lan  = lane & 15;

    // ---- build A = tanh(pi_i + pj_j + bl) as bf16 in LDS (vectorized x4) ----
    const float* piRow = pi + (size_t)(b * SS + i) * HH;
    for (int idx4 = tid; idx4 < (SS * HH) / 4; idx4 += 256) {
        int j  = idx4 / (HH / 4);
        int k4 = (idx4 - j * (HH / 4)) * 4;
        float4 pv = *reinterpret_cast<const float4*>(piRow + k4);
        float4 qv = *reinterpret_cast<const float4*>(pj + (size_t)(b * SS + j) * HH + k4);
        float4 bv = *reinterpret_cast<const float4*>(bl + k4);
        v4bf pack;
        pack[0] = (bf16)tanhf(pv.x + qv.x + bv.x);
        pack[1] = (bf16)tanhf(pv.y + qv.y + bv.y);
        pack[2] = (bf16)tanhf(pv.z + qv.z + bv.z);
        pack[3] = (bf16)tanhf(pv.w + qv.w + bv.w);
        *reinterpret_cast<v4bf*>(As + (size_t)j * ASTR + k4) = pack;
    }
    __syncthreads();

    v8f relc = {};                              // 16x16 rel accumulator tile per wave

    for (int nc = 0; nc < HH / 128; ++nc) {     // 6 column chunks of W1r
        const int nbase = nc * 128;

        // kick off async staging of the first K=64 B tile for this chunk
        async_copy_b_tile(W1rT, BsAddr, nbase, 0, tid);

        float b1v[8];
        #pragma unroll
        for (int ct = 0; ct < 8; ++ct) b1v[ct] = b1r[nbase + ct * 16 + lan];

        v8f c[8];
        #pragma unroll
        for (int ct = 0; ct < 8; ++ct) c[ct] = (v8f){};

        for (int t2 = 0; t2 < HH / 64; ++t2) {  // 12 K-tiles of 64
            const int cur = t2 & 1;
            asm volatile("s_wait_asynccnt 0x0" ::: "memory");
            __syncthreads();                    // tile[cur] visible to all waves
            if (t2 + 1 < HH / 64)               // prefetch next tile into other buffer
                async_copy_b_tile(W1rT, BsAddr + (unsigned)((cur ^ 1) * BTILE_BYTES),
                                  nbase, (t2 + 1) * 64, tid);

            const bf16* tile = Bs + (size_t)cur * BTILE_ELEMS;
            #pragma unroll
            for (int kk = 0; kk < 64; kk += 32) {
                const int kb = t2 * 64 + kk;
                // A fragment: 16-bit A 16x32 layout (K 0-7,16-23 | 8-15,24-31)
                Frag16 fa;
                const bf16* ap = As + (size_t)(w * 16 + lan) * ASTR + kb + hs * 8;
                fa.h[0] = *reinterpret_cast<const v8bf*>(ap);
                fa.h[1] = *reinterpret_cast<const v8bf*>(ap + 16);
                #pragma unroll
                for (int ct = 0; ct < 8; ++ct) {
                    // B fragment from LDS tile (K 0-7,16-23 | 8-15,24-31)
                    Frag16 fb;
                    const bf16* bp = tile + (size_t)(ct * 16 + lan) * BROW + kk + hs * 8;
                    fb.h[0] = *reinterpret_cast<const v8bf*>(bp);
                    fb.h[1] = *reinterpret_cast<const v8bf*>(bp + 16);
                    c[ct] = __builtin_amdgcn_wmma_f32_16x16x32_bf16(
                        false, fa.v, false, fb.v, (short)0, c[ct], false, false);
                }
            }
        }

        // epilogue: +b1r, mish, stash bf16 into A2 (rows of this wave's strip)
        #pragma unroll
        for (int ct = 0; ct < 8; ++ct) {
            #pragma unroll
            for (int v = 0; v < 8; ++v) {
                int j  = w * 16 + v + hs * 8;
                int nl = ct * 16 + lan;
                A2[(size_t)j * A2STR + nl] = (bf16)mishf(c[ct][v] + b1v[ct]);
            }
        }
        __syncthreads();

        // rel tile += A2(16x128 strip) @ W2rT(128x16 chunk)
        #pragma unroll
        for (int ks = 0; ks < 4; ++ks) {
            Frag16 fa2;
            const bf16* a2p = A2 + (size_t)(w * 16 + lan) * A2STR + ks * 32 + hs * 8;
            fa2.h[0] = *reinterpret_cast<const v8bf*>(a2p);
            fa2.h[1] = *reinterpret_cast<const v8bf*>(a2p + 16);
            Frag16 fb2;
            const bf16* b2p = W2rT + (size_t)lan * HH + nbase + ks * 32 + hs * 8;
            fb2.h[0] = *reinterpret_cast<const v8bf*>(b2p);
            fb2.h[1] = *reinterpret_cast<const v8bf*>(b2p + 16);
            relc = __builtin_amdgcn_wmma_f32_16x16x32_bf16(
                false, fa2.v, false, fb2.v, (short)0, relc, false, false);
        }
        __syncthreads();
    }

    // dump rel tiles to LDS, then per-row log_softmax over 13
    #pragma unroll
    for (int v = 0; v < 8; ++v)
        relbuf[(size_t)(w * 16 + v + hs * 8) * 16 + lan] = relc[v];
    __syncthreads();

    if (tid < SS) {
        int j = tid;
        float l[RR], mx = -1e30f;
        #pragma unroll
        for (int r = 0; r < RR; ++r) { l[r] = relbuf[j * 16 + r] + b2r[r]; mx = fmaxf(mx, l[r]); }
        float s = 0.f;
        #pragma unroll
        for (int r = 0; r < RR; ++r) s += __expf(l[r] - mx);
        float lse = mx + __logf(s);
        size_t off = ((size_t)b * (SS * SS) + (size_t)i * SS + j) * RR;
        #pragma unroll
        for (int r = 0; r < RR; ++r) rel_out[off + r] = l[r] - lse;
    }
}

// ---------------------------------------------------------------------------
extern "C" void kernel_launch(void* const* d_in, const int* in_sizes, int n_in,
                              void* d_out, int out_size, void* d_ws, size_t ws_size,
                              hipStream_t stream) {
    const float* emb = (const float*)d_in[0];
    const float* W1e = (const float*)d_in[1];
    const float* b1e = (const float*)d_in[2];
    const float* W2e = (const float*)d_in[3];
    const float* b2e = (const float*)d_in[4];
    const float* Wl  = (const float*)d_in[5];
    const float* bl  = (const float*)d_in[6];
    const float* W1r = (const float*)d_in[7];
    const float* b1r = (const float*)d_in[8];
    const float* W2r = (const float*)d_in[9];
    const float* b2r = (const float*)d_in[10];

    char* ws = (char*)d_ws;
    float* pi     = (float*)(ws);                          // 1024*768 f32
    float* pj     = (float*)(ws + 3145728);                // 1024*768 f32
    float* entlog = (float*)(ws + 6291456);                // 1024*4 f32
    bf16*  W1rT   = (bf16*)(ws + 6307840);                 // 768*768 bf16
    bf16*  W2rT   = (bf16*)(ws + 7487488);                 // 16*768 bf16

    float* entity = (float*)d_out;                         // 8*128*4
    float* rel    = (float*)d_out + BB * SS * EE;          // 8*16384*13

    entity_kernel<<<BB * SS, 256, 0, stream>>>(emb, W1e, b1e, W2e, b2e, entity, entlog);
    pij_kernel<<<dim3(BB * SS, 2), 256, 0, stream>>>(emb, entlog, Wl, pi, pj);
    conv_kernel<<<(HH * HH + 16 * HH + 255) / 256, 256, 0, stream>>>(W1r, W2r, W1rT, W2rT);

    size_t smem = SMEM_TOTAL;  // 278528 B (< 320 KB per WGP)
    hipFuncSetAttribute((const void*)rel_kernel,
                        hipFuncAttributeMaxDynamicSharedMemorySize, (int)smem);
    rel_kernel<<<BB * SS, 256, smem, stream>>>(pi, pj, bl, W1rT, b1r, W2rT, b2r, rel);
}
